// VoxelPointnetBackBone8x_13932873908760
// MI455X (gfx1250) — matrix operand order repaired and checked by reading, not compiled, not run on hardware
//
#include <hip/hip_runtime.h>
#include <hip/hip_bf16.h>
#include <math.h>

#define N_POINTS 32768
#define NUM_KP   4096
#define SA_TILE  256
#define FP_TILE  256

typedef __attribute__((ext_vector_type(16))) _Float16 v16h;
typedef __attribute__((ext_vector_type(8)))  float    v8f;

// ---------------------------------------------------------------------------
// Kernel 1: Furthest-point sampling. Single persistent 1024-thread block
// (32 wave32s). Each thread owns 32 points; mind2 lives in registers.
// Per iteration: update mind2 vs last pick, two-level argmax (shfl + LDS).
// ---------------------------------------------------------------------------
__global__ __launch_bounds__(1024) void k_fps(const float* __restrict__ xyz,
                                              float* __restrict__ kp_xyz) {
  __shared__ float s_lx, s_ly, s_lz;
  __shared__ float s_bv[32];
  __shared__ int   s_bi[32];
  const int t = threadIdx.x;
  const int lane = t & 31, wv = t >> 5;

  float mind2[32];
#pragma unroll
  for (int i = 0; i < 32; ++i) mind2[i] = 1e10f;

  if (t == 0) {
    s_lx = xyz[0]; s_ly = xyz[1]; s_lz = xyz[2];
    kp_xyz[0] = xyz[0]; kp_xyz[1] = xyz[1]; kp_xyz[2] = xyz[2];
  }

  for (int m = 1; m < NUM_KP; ++m) {
    __syncthreads();
    const float lx = s_lx, ly = s_ly, lz = s_lz;
    float bv = -1.0f; int bi = 0;
#pragma unroll
    for (int i = 0; i < 32; ++i) {
      const int p = t + i * 1024;
      const float dx = xyz[3 * p + 0] - lx;
      const float dy = xyz[3 * p + 1] - ly;
      const float dz = xyz[3 * p + 2] - lz;
      const float d2 = dx * dx + dy * dy + dz * dz;
      const float md = fminf(mind2[i], d2);
      mind2[i] = md;
      if (md > bv || (md == bv && p < bi)) { bv = md; bi = p; }
    }
    // wave-level argmax
#pragma unroll
    for (int off = 16; off >= 1; off >>= 1) {
      const float ov = __shfl_xor(bv, off, 32);
      const int   oi = __shfl_xor(bi, off, 32);
      if (ov > bv || (ov == bv && oi < bi)) { bv = ov; bi = oi; }
    }
    if (lane == 0) { s_bv[wv] = bv; s_bi[wv] = bi; }
    __syncthreads();
    if (wv == 0) {
      bv = s_bv[lane]; bi = s_bi[lane];
#pragma unroll
      for (int off = 16; off >= 1; off >>= 1) {
        const float ov = __shfl_xor(bv, off, 32);
        const int   oi = __shfl_xor(bi, off, 32);
        if (ov > bv || (ov == bv && oi < bi)) { bv = ov; bi = oi; }
      }
      if (lane == 0) {
        const float nx = xyz[3 * bi + 0], ny = xyz[3 * bi + 1], nz = xyz[3 * bi + 2];
        s_lx = nx; s_ly = ny; s_lz = nz;
        kp_xyz[3 * m + 0] = nx; kp_xyz[3 * m + 1] = ny; kp_xyz[3 * m + 2] = nz;
      }
    }
  }
}

// ---------------------------------------------------------------------------
// SA per-radius MLP: one wave handles one keypoint's 16 samples.
// Layer1: A[16 samples x K32 (5 used)] x W1[K x 16]  via v_wmma_f32_16x16x32_f16
// Layer2: A[16 samples x K32 (16 used)] x W2         via v_wmma_f32_16x16x32_f16
// A layout (16-bit, 16x32): lanes0-15 hold K{0..7,16..23}, lanes16-31 K{8..15,24..31}
// B as built here (32x16): lanes0-15 element h = W row h (N=lane, zero-padded), lanes16-31 zero
// D layout (f32 16x16): VGPR r -> M=r (lanes0-15, N=lane) / M=8+r (lanes16-31)
// ---------------------------------------------------------------------------
__device__ __forceinline__ void sa_mlp_radius(
    const float* __restrict__ xyz, const float* __restrict__ feats,
    float kx, float ky, float kz,
    const float* nnd, const int* nni,   // this wave's sorted nearest-16 (LDS)
    float* hbuf,                        // this wave's 16x16 layer-1 staging (LDS)
    const float* __restrict__ W1, const float* __restrict__ g1, const float* __restrict__ b1,
    const float* __restrict__ W2, const float* __restrict__ g2, const float* __restrict__ b2,
    float r2, int lane, float* __restrict__ out16) {
  const int m = lane & 15;            // sample row this lane feeds
  const int n = lane & 15;            // output column this lane owns
  const float dm = nnd[m];
  const int   im = nni[m];
  const float d0 = nnd[0];
  const int   i0 = nni[0];
  const bool empty  = !(i0 >= 0 && d0 <= r2);
  const bool validm = (im >= 0) && (dm <= r2);
  const int gi = validm ? im : (empty ? 0 : i0);   // ball_query fill semantics

  // ---- layer 1 ----
  v16h A1;
#pragma unroll
  for (int h = 0; h < 16; ++h) A1[h] = (_Float16)0.0f;
  if (lane < 16) {                    // channels K=0..4 live at h=0..4 in low lanes
    A1[0] = (_Float16)(xyz[3 * gi + 0] - kx);
    A1[1] = (_Float16)(xyz[3 * gi + 1] - ky);
    A1[2] = (_Float16)(xyz[3 * gi + 2] - kz);
    A1[3] = (_Float16)feats[2 * gi + 0];
    A1[4] = (_Float16)feats[2 * gi + 1];
  }
  v16h B1;
#pragma unroll
  for (int h = 0; h < 16; ++h) B1[h] = (_Float16)0.0f;
  if (lane < 16) {
#pragma unroll
    for (int h = 0; h < 5; ++h) B1[h] = (_Float16)W1[h * 16 + n];
  }
  v8f c1;
#pragma unroll
  for (int r = 0; r < 8; ++r) c1[r] = 0.0f;
  c1 = __builtin_amdgcn_wmma_f32_16x16x32_f16(false, A1, false, B1, (short)0, c1,
                                              false, false);
  const float g1v = g1[n], b1v = b1[n];
  {
    const int mbase = (lane < 16) ? 0 : 8;
#pragma unroll
    for (int r = 0; r < 8; ++r)
      hbuf[(mbase + r) * 16 + n] = fmaxf(c1[r] * g1v + b1v, 0.0f);
  }
  __syncthreads();

  // ---- layer 2 ----
  v16h A2;
#pragma unroll
  for (int h = 0; h < 16; ++h) A2[h] = (_Float16)0.0f;
  if (lane < 16) {                      // low lanes K=0..15, high lanes K=16..31 (unused)
#pragma unroll
    for (int h = 0; h < 16; ++h) A2[h] = (_Float16)hbuf[m * 16 + h];
  }
  v16h B2;
#pragma unroll
  for (int h = 0; h < 16; ++h) B2[h] = (_Float16)0.0f;
  if (lane < 16) {
#pragma unroll
    for (int h = 0; h < 16; ++h) B2[h] = (_Float16)W2[h * 16 + n];
  }
  v8f c2;
#pragma unroll
  for (int r = 0; r < 8; ++r) c2[r] = 0.0f;
  c2 = __builtin_amdgcn_wmma_f32_16x16x32_f16(false, A2, false, B2, (short)0, c2,
                                              false, false);
  const float g2v = g2[n], b2v = b2[n];
  float mx = -3.0e38f;
#pragma unroll
  for (int r = 0; r < 8; ++r) mx = fmaxf(mx, fmaxf(c2[r] * g2v + b2v, 0.0f));
  mx = fmaxf(mx, __shfl_xor(mx, 16, 32));          // combine M halves -> pool over 16 samples
  if (empty) mx = 0.0f;
  if (lane < 16) out16[n] = mx;
  __syncthreads();
}

// ---------------------------------------------------------------------------
// Kernel 2: SA layer. One wave per keypoint (8 waves / 256-thread block).
// Shared nearest-16 search (LDS-tiled points, per-lane sorted lists, wave
// merge via shuffles) serves both radii; then 4 WMMAs per keypoint.
// ---------------------------------------------------------------------------
__global__ __launch_bounds__(256) void k_sa(
    const float* __restrict__ xyz, const float* __restrict__ feats,
    const float* __restrict__ kp_xyz,
    const float* __restrict__ W1a, const float* __restrict__ g1a, const float* __restrict__ b1a,
    const float* __restrict__ W2a, const float* __restrict__ g2a, const float* __restrict__ b2a,
    const float* __restrict__ W1b, const float* __restrict__ g1b, const float* __restrict__ b1b,
    const float* __restrict__ W2b, const float* __restrict__ g2b, const float* __restrict__ b2b,
    float* __restrict__ pooled) {
  __shared__ float s_tx[SA_TILE], s_ty[SA_TILE], s_tz[SA_TILE];
  __shared__ float s_nnd[8][16];
  __shared__ int   s_nni[8][16];
  __shared__ float s_h[8][256];

  const int t = threadIdx.x, lane = t & 31, wv = t >> 5;
  const int kp = blockIdx.x * 8 + wv;
  const float kx = kp_xyz[3 * kp + 0];
  const float ky = kp_xyz[3 * kp + 1];
  const float kz = kp_xyz[3 * kp + 2];

  // per-lane sorted (ascending d2) top-16 candidate list
  float ld[16]; int li[16];
#pragma unroll
  for (int j = 0; j < 16; ++j) { ld[j] = 3.0e38f; li[j] = -1; }

  for (int tile = 0; tile < N_POINTS / SA_TILE; ++tile) {
    __syncthreads();
    const int gp = tile * SA_TILE + t;
    s_tx[t] = xyz[3 * gp + 0];
    s_ty[t] = xyz[3 * gp + 1];
    s_tz[t] = xyz[3 * gp + 2];
    __syncthreads();
#pragma unroll
    for (int k = 0; k < SA_TILE / 32; ++k) {
      const int j = lane + k * 32;
      const float dx = s_tx[j] - kx, dy = s_ty[j] - ky, dz = s_tz[j] - kz;
      const float d2 = dx * dx + dy * dy + dz * dz;
      if (d2 < ld[15]) {
        float v = d2; int ii = tile * SA_TILE + j;
#pragma unroll
        for (int q = 0; q < 16; ++q) {
          if (v < ld[q] || (v == ld[q] && ii < li[q])) {
            const float tv = ld[q]; const int ti = li[q];
            ld[q] = v; li[q] = ii; v = tv; ii = ti;
          }
        }
      }
    }
  }

  // merge 32 per-lane sorted lists into this wave's global nearest-16
  float rv[16]; int ri[16];
#pragma unroll
  for (int r = 0; r < 16; ++r) {
    float v = ld[0]; int ii = li[0]; int src = lane;
#pragma unroll
    for (int off = 16; off >= 1; off >>= 1) {
      const float ov = __shfl_xor(v, off, 32);
      const int   oi = __shfl_xor(ii, off, 32);
      const int   os = __shfl_xor(src, off, 32);
      if (ov < v || (ov == v && oi < ii)) { v = ov; ii = oi; src = os; }
    }
    rv[r] = v; ri[r] = ii;
    if (lane == src) {   // pop the consumed head (static-index shift)
#pragma unroll
      for (int q = 0; q < 15; ++q) { ld[q] = ld[q + 1]; li[q] = li[q + 1]; }
      ld[15] = 3.0e38f; li[15] = -1;
    }
  }
  if (lane == 0) {
#pragma unroll
    for (int q = 0; q < 16; ++q) { s_nnd[wv][q] = rv[q]; s_nni[wv][q] = ri[q]; }
  }
  __syncthreads();

  float* out = pooled + kp * 32;
  sa_mlp_radius(xyz, feats, kx, ky, kz, &s_nnd[wv][0], &s_nni[wv][0], &s_h[wv][0],
                W1a, g1a, b1a, W2a, g2a, b2a, 0.4f * 0.4f, lane, out);        // fa -> [0..15]
  sa_mlp_radius(xyz, feats, kx, ky, kz, &s_nnd[wv][0], &s_nni[wv][0], &s_h[wv][0],
                W1b, g1b, b1b, W2b, g2b, b2b, 0.8f * 0.8f, lane, out + 16);   // fb -> [16..31]
}

// ---------------------------------------------------------------------------
// Kernel 3: FP layer. Thread-per-point 3-NN over keypoints (LDS-tiled),
// inverse-distance interp to [32] staged in LDS, then [N,32]@Wf[32,16]
// via v_wmma_f32_16x16x32_f16 (16 points per WMMA, K=32 exact).
// ---------------------------------------------------------------------------
__global__ __launch_bounds__(256) void k_fp(
    const float* __restrict__ xyz, const float* __restrict__ kp_xyz,
    const float* __restrict__ pooled,
    const float* __restrict__ Wf, const float* __restrict__ gf,
    const float* __restrict__ bf, float* __restrict__ out) {
  __shared__ float s_kx[FP_TILE], s_ky[FP_TILE], s_kz[FP_TILE];
  __shared__ float s_fi[256 * 32];

  const int t = threadIdx.x, lane = t & 31, wv = t >> 5;
  const int p = blockIdx.x * 256 + t;
  const float px = xyz[3 * p + 0], py = xyz[3 * p + 1], pz = xyz[3 * p + 2];

  float t3d[3] = {3.0e38f, 3.0e38f, 3.0e38f};
  int   t3i[3] = {0, 0, 0};

  for (int tile = 0; tile < NUM_KP / FP_TILE; ++tile) {
    __syncthreads();
    const int gk = tile * FP_TILE + t;
    s_kx[t] = kp_xyz[3 * gk + 0];
    s_ky[t] = kp_xyz[3 * gk + 1];
    s_kz[t] = kp_xyz[3 * gk + 2];
    __syncthreads();
#pragma unroll 8
    for (int j = 0; j < FP_TILE; ++j) {
      const float dx = s_kx[j] - px, dy = s_ky[j] - py, dz = s_kz[j] - pz;
      float d2 = dx * dx + dy * dy + dz * dz;
      if (d2 < t3d[2]) {
        int ii = tile * FP_TILE + j;
#pragma unroll
        for (int q = 0; q < 3; ++q) {
          if (d2 < t3d[q]) {
            const float tv = t3d[q]; const int ti = t3i[q];
            t3d[q] = d2; t3i[q] = ii; d2 = tv; ii = ti;
          }
        }
      }
    }
  }

  float w0 = 1.0f / (sqrtf(fmaxf(t3d[0], 1e-12f)) + 1e-8f);
  float w1 = 1.0f / (sqrtf(fmaxf(t3d[1], 1e-12f)) + 1e-8f);
  float w2 = 1.0f / (sqrtf(fmaxf(t3d[2], 1e-12f)) + 1e-8f);
  const float wsum = w0 + w1 + w2;
  w0 /= wsum; w1 /= wsum; w2 /= wsum;
  const float* p0 = pooled + t3i[0] * 32;
  const float* p1 = pooled + t3i[1] * 32;
  const float* p2 = pooled + t3i[2] * 32;
#pragma unroll
  for (int c = 0; c < 32; ++c)
    s_fi[t * 32 + c] = w0 * p0[c] + w1 * p1[c] + w2 * p2[c];
  __syncthreads();

  // final GEMM: B = Wf (32x16), shared across both 16-point groups of this wave
  const int n = lane & 15;
  v16h B;
#pragma unroll
  for (int h = 0; h < 16; ++h) {
    const int K = ((h < 8) ? h : (h + 8)) + ((lane < 16) ? 0 : 8);
    B[h] = (_Float16)Wf[K * 16 + n];
  }
  const float gv = gf[n], bv = bf[n];

#pragma unroll
  for (int g = 0; g < 2; ++g) {
    const int m = lane & 15;
    const int rowt = wv * 32 + g * 16 + m;   // thread-in-block whose interp is row m
    v16h A;
#pragma unroll
    for (int h = 0; h < 16; ++h) {
      const int K = ((h < 8) ? h : (h + 8)) + ((lane < 16) ? 0 : 8);
      A[h] = (_Float16)s_fi[rowt * 32 + K];
    }
    v8f c;
#pragma unroll
    for (int r = 0; r < 8; ++r) c[r] = 0.0f;
    c = __builtin_amdgcn_wmma_f32_16x16x32_f16(false, A, false, B, (short)0, c,
                                               false, false);
    const int mbase = (lane < 16) ? 0 : 8;
#pragma unroll
    for (int r = 0; r < 8; ++r) {
      const float val = fmaxf(c[r] * gv + bv, 0.0f);
      const int pt = blockIdx.x * 256 + wv * 32 + g * 16 + mbase + r;
      out[pt * 16 + n] = val;
    }
  }
}

// ---------------------------------------------------------------------------
extern "C" void kernel_launch(void* const* d_in, const int* in_sizes, int n_in,
                              void* d_out, int out_size, void* d_ws, size_t ws_size,
                              hipStream_t stream) {
  (void)in_sizes; (void)n_in; (void)out_size; (void)ws_size;
  const float* xyz   = (const float*)d_in[0];
  const float* feats = (const float*)d_in[1];
  const float* Wf    = (const float*)d_in[2];
  const float* gf    = (const float*)d_in[3];
  const float* bf    = (const float*)d_in[4];
  const float* W1a   = (const float*)d_in[5];
  const float* g1a   = (const float*)d_in[6];
  const float* b1a   = (const float*)d_in[7];
  const float* W2a   = (const float*)d_in[8];
  const float* g2a   = (const float*)d_in[9];
  const float* b2a   = (const float*)d_in[10];
  const float* W1b   = (const float*)d_in[11];
  const float* g1b   = (const float*)d_in[12];
  const float* b1b   = (const float*)d_in[13];
  const float* W2b   = (const float*)d_in[14];
  const float* g2b   = (const float*)d_in[15];
  const float* b2b   = (const float*)d_in[16];
  float* out = (float*)d_out;

  float* kp_xyz = (float*)d_ws;            // NUM_KP * 3 floats
  float* pooled = kp_xyz + NUM_KP * 3;     // NUM_KP * 32 floats

  k_fps<<<1, 1024, 0, stream>>>(xyz, kp_xyz);
  k_sa<<<NUM_KP / 8, 256, 0, stream>>>(xyz, feats, kp_xyz,
                                       W1a, g1a, b1a, W2a, g2a, b2a,
                                       W1b, g1b, b1b, W2b, g2b, b2b, pooled);
  k_fp<<<N_POINTS / 256, 256, 0, stream>>>(xyz, kp_xyz, pooled, Wf, gf, bf, out);
}
